// ClusteredAttention_73581379715367
// MI455X (gfx1250) — matrix operand order repaired and checked by reading, not compiled, not run on hardware
//
#include <hip/hip_runtime.h>

// ---------------------------------------------------------------------------
// Clustered flash attention for MI455X (gfx1250, wave32, WMMA bf16->f32).
// b=4, l=4096, d=64. One 128-thread block (4 waves) handles 64 query rows of
// one batch; each wave owns 16 rows. Column loop streams 64-key tiles through
// LDS (K row-major bf16, V transposed bf16), online softmax in registers.
// This revision batches all LDS fragment loads ahead of each WMMA group so a
// single s_wait_dscnt covers the group and the 8 WMMAs issue back-to-back.
// ---------------------------------------------------------------------------

#define BATCH  4
#define SEQ    4096
#define DMODEL 64
#define CTILE  64
#define NWAVES 4
#define NTHREADS 128

typedef __attribute__((ext_vector_type(16))) __bf16 v16bf;
typedef __attribute__((ext_vector_type(8)))  float  v8f;
typedef __attribute__((ext_vector_type(4)))  unsigned int u32x4;

union BF16Frag { v16bf v; u32x4 u[2]; };

// B-fragment: lane holds col N = lane%16, K = {o..o+7} U {o+16..o+23}, o=8*(lane/16).
// Caller passes (row pointer + 8*(lane/16)).
__device__ __forceinline__ v16bf lds_bfrag(const __bf16* p) {
  BF16Frag f;
  f.u[0] = *(const u32x4*)(p);
  f.u[1] = *(const u32x4*)(p + 16);
  return f.v;
}

// A-fragment: lane holds row M = lane%16, K = {o..o+7} U {o+16..o+23}, o=8*(lane/16).
// Caller passes row pointer (+ kc*32) and o.
__device__ __forceinline__ v16bf lds_afrag(const __bf16* rowp, int o) {
  BF16Frag f;
  f.u[0] = *(const u32x4*)(rowp + o);
  f.u[1] = *(const u32x4*)(rowp + o + 16);
  return f.v;
}

// A-fragment built from global fp32 (Q): convert 16 floats -> bf16.
__device__ __forceinline__ v16bf q_afrag(const float* rowp, int base) {
  float4 f0 = *(const float4*)(rowp + base);
  float4 f1 = *(const float4*)(rowp + base + 4);
  float4 f2 = *(const float4*)(rowp + base + 16);
  float4 f3 = *(const float4*)(rowp + base + 20);
  v16bf v;
  v[0]=(__bf16)f0.x;  v[1]=(__bf16)f0.y;  v[2]=(__bf16)f0.z;  v[3]=(__bf16)f0.w;
  v[4]=(__bf16)f1.x;  v[5]=(__bf16)f1.y;  v[6]=(__bf16)f1.z;  v[7]=(__bf16)f1.w;
  v[8]=(__bf16)f2.x;  v[9]=(__bf16)f2.y;  v[10]=(__bf16)f2.z; v[11]=(__bf16)f2.w;
  v[12]=(__bf16)f3.x; v[13]=(__bf16)f3.y; v[14]=(__bf16)f3.z; v[15]=(__bf16)f3.w;
  return v;
}

__global__ __launch_bounds__(NTHREADS)
void clustered_attn_fa(const float* __restrict__ Qg, const float* __restrict__ Kg,
                       const float* __restrict__ Vg, const int* __restrict__ LAB,
                       float* __restrict__ OUT) {
  __shared__ __bf16 Ksh[CTILE][DMODEL];      // key tile, row-major        (8 KB)
  __shared__ __bf16 VTsh[DMODEL][CTILE];     // value tile, transposed     (8 KB)
  __shared__ __bf16 Psh[NWAVES][16 * CTILE]; // per-wave probability tile  (8 KB)
  __shared__ int    labsh[CTILE];

  const int tid  = threadIdx.x;
  const int lane = tid & 31;
  const int wv   = tid >> 5;
  const int l15  = lane & 15;
  const int hi   = lane >> 4;
  const int oA   = hi * 8;    // A-fragment K-offset base
  const int oB   = hi * 16;   // unused; B-fragments also use oA

  const int tiles = SEQ / CTILE;
  const int batch = blockIdx.x / tiles;
  const int q0    = (blockIdx.x % tiles) * CTILE;
  const int qw    = q0 + wv * 16;

  // scale = 1/sqrt(d); fold in log2(e) so we can use exp2 (v_exp_f32).
  const float SCL = 0.125f * 1.44269504088896340736f;

  // ---- Q fragments (resident for whole kernel) ----
  const float* qrow = Qg + ((size_t)(batch * SEQ + qw + l15)) * DMODEL;
  const v16bf qf0 = q_afrag(qrow, 0 * 32 + oA);
  const v16bf qf1 = q_afrag(qrow, 1 * 32 + oA);

  // ---- per-row labels (C-layout: row m = v + 8*hi) ----
  int rlab[8];
#pragma unroll
  for (int v = 0; v < 8; ++v) {
    const int row = qw + v + 8 * hi;
    rlab[v] = (row < SEQ - 1) ? LAB[batch * (SEQ - 1) + row] : -1;
  }

  const v8f vzero = {0.f,0.f,0.f,0.f,0.f,0.f,0.f,0.f};
  v8f accO[4] = {vzero, vzero, vzero, vzero};
  float mrun[8], srun[8];
#pragma unroll
  for (int v = 0; v < 8; ++v) { mrun[v] = -1.0e30f; srun[v] = 0.f; }

  for (int t = 0; t < tiles; ++t) {
    const int col0 = t * CTILE;
    __syncthreads();  // previous iteration's LDS consumers done

    // ---- cooperative K/V tile load: 2 threads per row, 32 d each ----
    {
      const int r = tid >> 1;
      const int dbase = (tid & 1) * 32;
      const size_t rowoff = ((size_t)(batch * SEQ + col0 + r)) * DMODEL + dbase;
      const float* krow = Kg + rowoff;
      const float* vrow = Vg + rowoff;
#pragma unroll
      for (int i = 0; i < 8; ++i) {
        const int dc = dbase + 4 * i;
        float4 kf = *(const float4*)(krow + 4 * i);
        union { __bf16 h[4]; unsigned long long u; } pk;
        pk.h[0] = (__bf16)kf.x; pk.h[1] = (__bf16)kf.y;
        pk.h[2] = (__bf16)kf.z; pk.h[3] = (__bf16)kf.w;
        *(unsigned long long*)(&Ksh[r][0] + dc) = pk.u;   // ds_store_b64
        float4 vf = *(const float4*)(vrow + 4 * i);
        VTsh[dc + 0][r] = (__bf16)vf.x;                   // transposed scatter
        VTsh[dc + 1][r] = (__bf16)vf.y;
        VTsh[dc + 2][r] = (__bf16)vf.z;
        VTsh[dc + 3][r] = (__bf16)vf.w;
      }
      if (tid < CTILE) {
        const int c = col0 + tid;
        labsh[tid] = (c < SEQ - 1) ? LAB[batch * (SEQ - 1) + c] : -1;
      }
    }
    __syncthreads();

    // prefetch next tile into cache hierarchy (global_prefetch_b8)
    if (t + 1 < tiles) {
      const size_t noff = ((size_t)(batch * SEQ + col0 + CTILE + (tid >> 1))) * DMODEL
                          + (size_t)((tid & 1) * 32);
      __builtin_prefetch(Kg + noff, 0, 3);
      __builtin_prefetch(Vg + noff, 0, 3);
    }

    // ---- S = Q * K^T : preload all 8 B-fragments, then 8 back-to-back WMMAs
    v16bf kb[8];
#pragma unroll
    for (int nt = 0; nt < 4; ++nt) {
      const __bf16* krow = &Ksh[nt * 16 + l15][0];
      kb[nt * 2 + 0] = lds_bfrag(krow + 0  + oA);
      kb[nt * 2 + 1] = lds_bfrag(krow + 32 + oA);
    }
    v8f accS[4];
#pragma unroll
    for (int nt = 0; nt < 4; ++nt) {
      v8f a = __builtin_amdgcn_wmma_f32_16x16x32_bf16(false, qf0, false, kb[nt * 2 + 0],
                                                      (short)0, vzero, false, false);
      accS[nt] = __builtin_amdgcn_wmma_f32_16x16x32_bf16(false, qf1, false, kb[nt * 2 + 1],
                                                         (short)0, a, false, false);
    }

    // ---- mask + scale into sv ----
    float sv[4][8];
#pragma unroll
    for (int nt = 0; nt < 4; ++nt) {
      const int clab = labsh[nt * 16 + l15];
#pragma unroll
      for (int v = 0; v < 8; ++v) {
        const float s = accS[nt][v] * SCL;
        const bool msk = (rlab[v] != clab) && (rlab[v] >= 0) && (clab >= 0);
        sv[nt][v] = msk ? -1.0e30f : s;
      }
    }

    // ---- online softmax update (per row m = v + 8*hi) ----
#pragma unroll
    for (int v = 0; v < 8; ++v) {
      float mt = fmaxf(fmaxf(sv[0][v], sv[1][v]), fmaxf(sv[2][v], sv[3][v]));
#pragma unroll
      for (int off = 8; off >= 1; off >>= 1)
        mt = fmaxf(mt, __shfl_xor(mt, off, 32));
      const float mnew = fmaxf(mrun[v], mt);
      const float corr = exp2f(mrun[v] - mnew);
      mrun[v] = mnew;
      srun[v] *= corr;
#pragma unroll
      for (int dt = 0; dt < 4; ++dt) accO[dt][v] *= corr;
      float ps = 0.f;
#pragma unroll
      for (int nt = 0; nt < 4; ++nt) {
        const float p = exp2f(sv[nt][v] - mnew);
        sv[nt][v] = p;
        ps += p;
      }
#pragma unroll
      for (int off = 8; off >= 1; off >>= 1)
        ps += __shfl_xor(ps, off, 32);
      srun[v] += ps;
    }

    // ---- P (C-layout f32) -> per-wave LDS (row-major bf16) ----
    __bf16* pb = &Psh[wv][0];
#pragma unroll
    for (int nt = 0; nt < 4; ++nt)
#pragma unroll
      for (int v = 0; v < 8; ++v)
        pb[(v + 8 * hi) * CTILE + nt * 16 + l15] = (__bf16)sv[nt][v];
    // same-wave LDS is in-order; fence compiler + wait for stores to land
    asm volatile("s_wait_dscnt 0x0" ::: "memory");

    // ---- O += P * V : preload P A-frags and all 8 V^T B-frags, then 8 WMMAs
    const __bf16* prow = pb + l15 * CTILE;
    const v16bf p0 = lds_afrag(prow + 0,  oA);
    const v16bf p1 = lds_afrag(prow + 32, oA);
    v16bf vb[8];
#pragma unroll
    for (int dt = 0; dt < 4; ++dt) {
      const __bf16* vtrow = &VTsh[dt * 16 + l15][0];
      vb[dt * 2 + 0] = lds_bfrag(vtrow + 0  + oA);
      vb[dt * 2 + 1] = lds_bfrag(vtrow + 32 + oA);
    }
#pragma unroll
    for (int dt = 0; dt < 4; ++dt) {
      accO[dt] = __builtin_amdgcn_wmma_f32_16x16x32_bf16(false, p0, false, vb[dt * 2 + 0],
                                                         (short)0, accO[dt], false, false);
      accO[dt] = __builtin_amdgcn_wmma_f32_16x16x32_bf16(false, p1, false, vb[dt * 2 + 1],
                                                         (short)0, accO[dt], false, false);
    }
  }

  // ---- normalize and store (C-layout: row v+8*hi, col dt*16+l15) ----
  float* obase = OUT + ((size_t)(batch * SEQ + qw)) * DMODEL;
#pragma unroll
  for (int v = 0; v < 8; ++v) {
    const float inv = 1.0f / srun[v];
    float* orow = obase + (size_t)(v + 8 * hi) * DMODEL + l15;
#pragma unroll
    for (int dt = 0; dt < 4; ++dt)
      orow[dt * 16] = accO[dt][v] * inv;
  }
}

extern "C" void kernel_launch(void* const* d_in, const int* in_sizes, int n_in,
                              void* d_out, int out_size, void* d_ws, size_t ws_size,
                              hipStream_t stream) {
  (void)in_sizes; (void)n_in; (void)out_size; (void)d_ws; (void)ws_size;
  const float* q   = (const float*)d_in[0];
  const float* k   = (const float*)d_in[1];
  const float* v   = (const float*)d_in[2];
  const int*   lab = (const int*)d_in[3];
  float* out = (float*)d_out;

  dim3 grid(BATCH * (SEQ / CTILE));   // 256 blocks
  dim3 block(NTHREADS);               // 4 waves of 32
  hipLaunchKernelGGL(clustered_attn_fa, grid, block, 0, stream,
                     q, k, v, lab, out);
}
